// SelfAttention_28587302322341
// MI455X (gfx1250) — matrix operand order repaired and checked by reading, not compiled, not run on hardware
//
#include <hip/hip_runtime.h>
#include <hip/hip_bf16.h>

// ---------------------------------------------------------------------------
// Self-attention (single head, D=1024) via bf16 WMMA on gfx1250 (wave32).
//   xb = bf16(x); Wb = bf16(W)  (one streaming conversion pass)
//   Q = xb Wq^T ; K = xb Wk^T ; V^T = (xb Wv^T)^T     (bf16, fp32 accum)
//   Sc = (Q K^T) / D^2  (fp32)
//   P  = softmax_row(Sc) (bf16)
//   out = P V  (fp32)
// All GEMMs: GLOBAL_LOAD_ASYNC_TO_LDS_B128 tile DMA (ASYNCcnt), ping-pong LDS
// double buffering with one barrier per K-step, padded LDS pitch (40 bf16)
// to avoid bank conflicts on ds_load_b128 fragment reads.
// ---------------------------------------------------------------------------

typedef __attribute__((ext_vector_type(16))) __bf16 v16bf;
typedef __attribute__((ext_vector_type(8)))  float  v8f;
typedef __attribute__((ext_vector_type(2)))  __bf16 v2bf;
typedef __attribute__((ext_vector_type(2)))  float  v2f;

#define B_   4
#define S_   2048
#define D_   1024
#define M_   (B_ * S_)      // 8192 tokens
#define SP_  40             // LDS tile pitch in bf16 (80B = 20-bank stride)

union FragU { unsigned u[8]; v16bf v; };

__device__ __forceinline__ unsigned pack_bf16(float a, float b) {
    v2f f = {a, b};
    v2bf h = __builtin_convertvector(f, v2bf);   // native v_cvt_pk path (RNE)
    union { v2bf h; unsigned u; } cv; cv.h = h;
    return cv.u;
}
__device__ __forceinline__ unsigned short f32_to_bf16(float f) {
    union { __bf16 h; unsigned short u; } cv;
    cv.h = (__bf16)f;
    return cv.u;
}

// A-matrix fragment (16x32 bf16, MxK), tile pitch SP_:
// lanes 0-15: VGPR0-3 = K 0..7, VGPR4-7 = K 16..23
// lanes16-31: VGPR0-3 = K 8..15, VGPR4-7 = K 24..31
__device__ __forceinline__ v16bf load_fragA(const __bf16* sh, int row, int hi) {
    const __bf16* base = sh + row * SP_;
    FragU f;
    *(uint4*)&f.u[0] = *(const uint4*)(base + 8 * hi);
    *(uint4*)&f.u[4] = *(const uint4*)(base + 16 + 8 * hi);
    return f.v;
}
// B-matrix fragment (32x16 bf16, KxN), tile stored N-major, K contiguous:
// lanes 0-15: VGPR0-3 = K 0..7, VGPR4-7 = K 16..23 ; lanes16-31: K 8..15, K 24..31
__device__ __forceinline__ v16bf load_fragB(const __bf16* sh, int row, int hi) {
    const __bf16* base = sh + row * SP_;
    FragU f;
    *(uint4*)&f.u[0] = *(const uint4*)(base + 8 * hi);
    *(uint4*)&f.u[4] = *(const uint4*)(base + 16 + 8 * hi);
    return f.v;
}

// ---- CDNA5 async global->LDS copy (tracked by ASYNCcnt) --------------------
__device__ __forceinline__ void async_copy_b128(unsigned lds_off, const __bf16* g) {
    asm volatile("global_load_async_to_lds_b128 %0, %1, off"
                 :: "v"(lds_off), "v"(g) : "memory");
}
__device__ __forceinline__ void wait_async0() {
    asm volatile("s_wait_asynccnt 0x0" ::: "memory");
}
__device__ __forceinline__ unsigned lds_off_of(const void* p) {
    return (unsigned)(size_t)p;   // low 32 bits of generic LDS pointer = LDS offset
}

// ---------------------------------------------------------------------------
// Kernel 0: streaming f32 -> bf16 conversion (8 elements / thread).
// ---------------------------------------------------------------------------
__global__ __launch_bounds__(256)
void cvt_bf16_kernel(const float* __restrict__ src, __bf16* __restrict__ dst)
{
    const size_t i = ((size_t)blockIdx.x * 256 + threadIdx.x) * 8;
    float4 a = *(const float4*)(src + i);
    float4 b = *(const float4*)(src + i + 4);
    uint4 t;
    t.x = pack_bf16(a.x, a.y); t.y = pack_bf16(a.z, a.w);
    t.z = pack_bf16(b.x, b.y); t.w = pack_bf16(b.z, b.w);
    *(uint4*)(dst + i) = t;
}

// ---------------------------------------------------------------------------
// Kernel 1: projection  C[M,D] (bf16) = Xb[M,D] * Wb[D,D]^T   (bf16 GEMM)
// Tile 128(M) x 64(N); waves on 4x2 grid -> each wave 32m x 32n.
// Async DMA tiles, ping-pong, 1 barrier/iter.
// transpose_out writes C^T[e][token] (V) so the PV GEMM stays K-contiguous.
// ---------------------------------------------------------------------------
__global__ __launch_bounds__(256)
void proj_kernel(const __bf16* __restrict__ Xb, const __bf16* __restrict__ Wb,
                 __bf16* __restrict__ C, int transpose_out)
{
    __shared__ __bf16 shA[2][128 * SP_];
    __shared__ __bf16 shB[2][64 * SP_];

    const int tid  = threadIdx.x;
    const int lane = tid & 31;
    const int wave = tid >> 5;
    const int lm   = lane & 15;
    const int lhi  = lane >> 4;
    const int wm   = wave & 3;      // 32-row m-subtile
    const int wn   = wave >> 2;     // 32-col n-subtile
    const int m0   = blockIdx.x * 128;
    const int n0   = blockIdx.y * 64;

    const int rowA = tid >> 1, cbA = (tid & 1) * 16;
    const int rowB = tid >> 2, cbB = (tid & 3) * 8;
    const __bf16* gA = Xb + (size_t)(m0 + rowA) * D_ + cbA;
    const __bf16* gB = Wb + (size_t)(n0 + rowB) * D_ + cbB;
    const unsigned la0 = lds_off_of(&shA[0][rowA * SP_ + cbA]);
    const unsigned la1 = lds_off_of(&shA[1][rowA * SP_ + cbA]);
    const unsigned lb0 = lds_off_of(&shB[0][rowB * SP_ + cbB]);
    const unsigned lb1 = lds_off_of(&shB[1][rowB * SP_ + cbB]);

    v8f acc[2][2];
#pragma unroll
    for (int i = 0; i < 2; ++i)
#pragma unroll
        for (int j = 0; j < 2; ++j)
#pragma unroll
            for (int r = 0; r < 8; ++r) acc[i][j][r] = 0.0f;

    async_copy_b128(la0,      gA);
    async_copy_b128(la0 + 16, gA + 8);
    async_copy_b128(lb0,      gB);

    const int NIT = D_ / 32;
    int p = 0;
    for (int it = 0; it < NIT; ++it) {
        wait_async0();
        __syncthreads();
        if (it + 1 < NIT) {
            const __bf16* nA = gA + (it + 1) * 32;
            const __bf16* nB = gB + (it + 1) * 32;
            const unsigned da = p ? la0 : la1;
            const unsigned db = p ? lb0 : lb1;
            async_copy_b128(da,      nA);
            async_copy_b128(da + 16, nA + 8);
            async_copy_b128(db,      nB);
        }
        const __bf16* sa = &shA[p][0];
        const __bf16* sb = &shB[p][0];
        v16bf a0 = load_fragA(sa, wm * 32 + lm, lhi);
        v16bf a1 = load_fragA(sa, wm * 32 + 16 + lm, lhi);
        v16bf b0 = load_fragB(sb, wn * 32 + lm, lhi);
        v16bf b1 = load_fragB(sb, wn * 32 + 16 + lm, lhi);
        acc[0][0] = __builtin_amdgcn_wmma_f32_16x16x32_bf16(
            false, a0, false, b0, (short)0, acc[0][0], false, false);
        acc[0][1] = __builtin_amdgcn_wmma_f32_16x16x32_bf16(
            false, a0, false, b1, (short)0, acc[0][1], false, false);
        acc[1][0] = __builtin_amdgcn_wmma_f32_16x16x32_bf16(
            false, a1, false, b0, (short)0, acc[1][0], false, false);
        acc[1][1] = __builtin_amdgcn_wmma_f32_16x16x32_bf16(
            false, a1, false, b1, (short)0, acc[1][1], false, false);
        p ^= 1;
    }

    if (!transpose_out) {
        unsigned short* Cs = (unsigned short*)C;
#pragma unroll
        for (int i = 0; i < 2; ++i) {
            const int rbase = m0 + wm * 32 + i * 16 + 8 * lhi;
#pragma unroll
            for (int j = 0; j < 2; ++j) {
                const int ncol = n0 + wn * 32 + j * 16 + lm;
#pragma unroll
                for (int r = 0; r < 8; ++r)
                    Cs[(size_t)(rbase + r) * D_ + ncol] = f32_to_bf16(acc[i][j][r]);
            }
        }
    } else {
#pragma unroll
        for (int i = 0; i < 2; ++i) {
            const int mb = m0 + wm * 32 + i * 16 + 8 * lhi;
#pragma unroll
            for (int j = 0; j < 2; ++j) {
                const int e = n0 + wn * 32 + j * 16 + lm;
                uint4 t;
                t.x = pack_bf16(acc[i][j][0], acc[i][j][1]);
                t.y = pack_bf16(acc[i][j][2], acc[i][j][3]);
                t.z = pack_bf16(acc[i][j][4], acc[i][j][5]);
                t.w = pack_bf16(acc[i][j][6], acc[i][j][7]);
                *(uint4*)(C + (size_t)e * M_ + mb) = t;
            }
        }
    }
}

// ---------------------------------------------------------------------------
// Kernel 2: scores  Sc[b][q][k] = scale * dot(Q[b,q,:], K[b,k,:])  (fp32)
// Tile 128(q) x 128(k); waves 4x2 -> each wave 32q x 64k (8 WMMAs / 6 frags).
// ---------------------------------------------------------------------------
__global__ __launch_bounds__(256)
void scores_kernel(const __bf16* __restrict__ Q, const __bf16* __restrict__ Km,
                   float* __restrict__ Sc, float scale)
{
    __shared__ __bf16 shQ[2][128 * SP_];
    __shared__ __bf16 shK[2][128 * SP_];

    const int tid  = threadIdx.x;
    const int lane = tid & 31;
    const int wave = tid >> 5;
    const int lm   = lane & 15;
    const int lhi  = lane >> 4;
    const int wq   = wave & 3;      // 32-row q-subtile
    const int wk   = wave >> 2;     // 64-col k-subtile
    const int b    = blockIdx.z;
    const int q0   = blockIdx.x * 128;
    const int k0   = blockIdx.y * 128;

    const int row = tid >> 1;
    const int cb  = (tid & 1) * 16;
    const __bf16* gq = Q  + ((size_t)b * S_ + q0 + row) * D_ + cb;
    const __bf16* gk = Km + ((size_t)b * S_ + k0 + row) * D_ + cb;
    const unsigned lq0 = lds_off_of(&shQ[0][row * SP_ + cb]);
    const unsigned lq1 = lds_off_of(&shQ[1][row * SP_ + cb]);
    const unsigned lk0 = lds_off_of(&shK[0][row * SP_ + cb]);
    const unsigned lk1 = lds_off_of(&shK[1][row * SP_ + cb]);

    v8f acc[2][4];
#pragma unroll
    for (int i = 0; i < 2; ++i)
#pragma unroll
        for (int j = 0; j < 4; ++j)
#pragma unroll
            for (int r = 0; r < 8; ++r) acc[i][j][r] = 0.0f;

    async_copy_b128(lq0,      gq);
    async_copy_b128(lq0 + 16, gq + 8);
    async_copy_b128(lk0,      gk);
    async_copy_b128(lk0 + 16, gk + 8);

    const int NIT = D_ / 32;
    int p = 0;
    for (int it = 0; it < NIT; ++it) {
        wait_async0();
        __syncthreads();
        if (it + 1 < NIT) {
            const __bf16* nq = gq + (it + 1) * 32;
            const __bf16* nk = gk + (it + 1) * 32;
            const unsigned dq = p ? lq0 : lq1;
            const unsigned dk = p ? lk0 : lk1;
            async_copy_b128(dq,      nq);
            async_copy_b128(dq + 16, nq + 8);
            async_copy_b128(dk,      nk);
            async_copy_b128(dk + 16, nk + 8);
        }
        const __bf16* sq = &shQ[p][0];
        const __bf16* sk = &shK[p][0];
        v16bf a0 = load_fragA(sq, wq * 32 + lm, lhi);
        v16bf a1 = load_fragA(sq, wq * 32 + 16 + lm, lhi);
        v16bf bfr[4];
#pragma unroll
        for (int j = 0; j < 4; ++j)
            bfr[j] = load_fragB(sk, wk * 64 + j * 16 + lm, lhi);
#pragma unroll
        for (int j = 0; j < 4; ++j)
            acc[0][j] = __builtin_amdgcn_wmma_f32_16x16x32_bf16(
                false, a0, false, bfr[j], (short)0, acc[0][j], false, false);
#pragma unroll
        for (int j = 0; j < 4; ++j)
            acc[1][j] = __builtin_amdgcn_wmma_f32_16x16x32_bf16(
                false, a1, false, bfr[j], (short)0, acc[1][j], false, false);
        p ^= 1;
    }

#pragma unroll
    for (int i = 0; i < 2; ++i) {
        const int rbase = q0 + wq * 32 + i * 16 + 8 * lhi;
#pragma unroll
        for (int j = 0; j < 4; ++j) {
            const int kcol = k0 + wk * 64 + j * 16 + lm;
#pragma unroll
            for (int r = 0; r < 8; ++r)
                Sc[((size_t)b * S_ + rbase + r) * S_ + kcol] = acc[i][j][r] * scale;
        }
    }
}

// ---------------------------------------------------------------------------
// Kernel 3: row softmax over 2048, one block (256 thr) per row. P is bf16.
// ---------------------------------------------------------------------------
__global__ __launch_bounds__(256)
void softmax_kernel(const float* __restrict__ Sc, __bf16* __restrict__ P)
{
    __shared__ float red[256];
    const int row = blockIdx.x;          // b*S + q
    const int tid = threadIdx.x;

    const float* src = Sc + (size_t)row * S_ + tid * 8;
    float4 v0 = *(const float4*)(src);
    float4 v1 = *(const float4*)(src + 4);

    float m = fmaxf(fmaxf(fmaxf(v0.x, v0.y), fmaxf(v0.z, v0.w)),
                    fmaxf(fmaxf(v1.x, v1.y), fmaxf(v1.z, v1.w)));
    red[tid] = m;
    __syncthreads();
#pragma unroll
    for (int s = 128; s > 0; s >>= 1) {
        if (tid < s) red[tid] = fmaxf(red[tid], red[tid + s]);
        __syncthreads();
    }
    m = red[0];
    __syncthreads();

    float e[8];
    e[0] = __expf(v0.x - m); e[1] = __expf(v0.y - m);
    e[2] = __expf(v0.z - m); e[3] = __expf(v0.w - m);
    e[4] = __expf(v1.x - m); e[5] = __expf(v1.y - m);
    e[6] = __expf(v1.z - m); e[7] = __expf(v1.w - m);
    float sum = (e[0] + e[1] + e[2] + e[3]) + (e[4] + e[5] + e[6] + e[7]);
    red[tid] = sum;
    __syncthreads();
#pragma unroll
    for (int s = 128; s > 0; s >>= 1) {
        if (tid < s) red[tid] += red[tid + s];
        __syncthreads();
    }
    const float inv = 1.0f / red[0];

    uint4 t;
    t.x = pack_bf16(e[0] * inv, e[1] * inv);
    t.y = pack_bf16(e[2] * inv, e[3] * inv);
    t.z = pack_bf16(e[4] * inv, e[5] * inv);
    t.w = pack_bf16(e[6] * inv, e[7] * inv);
    *(uint4*)(P + (size_t)row * S_ + tid * 8) = t;
}

// ---------------------------------------------------------------------------
// Kernel 4: context  out[b,q,d] = sum_k P[b,q,k] * V[b,k,d]   (fp32 out)
// V^T[d][token] keeps the B fragment K(token)-contiguous.
// Tile 128(q) x 64(d); waves 4x2 -> each 32q x 32d.
// ---------------------------------------------------------------------------
__global__ __launch_bounds__(256)
void context_kernel(const __bf16* __restrict__ P, const __bf16* __restrict__ Vt,
                    float* __restrict__ Out)
{
    __shared__ __bf16 shP[2][128 * SP_];
    __shared__ __bf16 shV[2][64 * SP_];

    const int tid  = threadIdx.x;
    const int lane = tid & 31;
    const int wave = tid >> 5;
    const int lm   = lane & 15;
    const int lhi  = lane >> 4;
    const int wq   = wave & 3;      // 32-row q-subtile
    const int wd   = wave >> 2;     // 32-col d-subtile
    const int b    = blockIdx.z;
    const int q0   = blockIdx.x * 128;
    const int d0   = blockIdx.y * 64;

    const int rowp = tid >> 1, cbp = (tid & 1) * 16;
    const int rowv = tid >> 2, cbv = (tid & 3) * 8;
    const __bf16* gp = P  + ((size_t)b * S_ + q0 + rowp) * S_ + cbp;
    const __bf16* gv = Vt + (size_t)(d0 + rowv) * M_ + (size_t)b * S_ + cbv;
    const unsigned lp0 = lds_off_of(&shP[0][rowp * SP_ + cbp]);
    const unsigned lp1 = lds_off_of(&shP[1][rowp * SP_ + cbp]);
    const unsigned lv0 = lds_off_of(&shV[0][rowv * SP_ + cbv]);
    const unsigned lv1 = lds_off_of(&shV[1][rowv * SP_ + cbv]);

    v8f acc[2][2];
#pragma unroll
    for (int i = 0; i < 2; ++i)
#pragma unroll
        for (int j = 0; j < 2; ++j)
#pragma unroll
            for (int r = 0; r < 8; ++r) acc[i][j][r] = 0.0f;

    async_copy_b128(lp0,      gp);
    async_copy_b128(lp0 + 16, gp + 8);
    async_copy_b128(lv0,      gv);

    const int NIT = S_ / 32;
    int p = 0;
    for (int it = 0; it < NIT; ++it) {
        wait_async0();
        __syncthreads();
        if (it + 1 < NIT) {
            const __bf16* np = gp + (it + 1) * 32;
            const __bf16* nv = gv + (it + 1) * 32;
            const unsigned dp = p ? lp0 : lp1;
            const unsigned dv = p ? lv0 : lv1;
            async_copy_b128(dp,      np);
            async_copy_b128(dp + 16, np + 8);
            async_copy_b128(dv,      nv);
        }
        const __bf16* sp = &shP[p][0];
        const __bf16* sv = &shV[p][0];
        v16bf a0 = load_fragA(sp, wq * 32 + lm, lhi);
        v16bf a1 = load_fragA(sp, wq * 32 + 16 + lm, lhi);
        v16bf b0 = load_fragB(sv, wd * 32 + lm, lhi);
        v16bf b1 = load_fragB(sv, wd * 32 + 16 + lm, lhi);
        acc[0][0] = __builtin_amdgcn_wmma_f32_16x16x32_bf16(
            false, a0, false, b0, (short)0, acc[0][0], false, false);
        acc[0][1] = __builtin_amdgcn_wmma_f32_16x16x32_bf16(
            false, a0, false, b1, (short)0, acc[0][1], false, false);
        acc[1][0] = __builtin_amdgcn_wmma_f32_16x16x32_bf16(
            false, a1, false, b0, (short)0, acc[1][0], false, false);
        acc[1][1] = __builtin_amdgcn_wmma_f32_16x16x32_bf16(
            false, a1, false, b1, (short)0, acc[1][1], false, false);
        p ^= 1;
    }

#pragma unroll
    for (int i = 0; i < 2; ++i) {
        const int rbase = q0 + wq * 32 + i * 16 + 8 * lhi;
#pragma unroll
        for (int j = 0; j < 2; ++j) {
            const int dcol = d0 + wd * 32 + j * 16 + lm;
#pragma unroll
            for (int r = 0; r < 8; ++r)
                Out[((size_t)b * S_ + rbase + r) * D_ + dcol] = acc[i][j][r];
        }
    }
}

// ---------------------------------------------------------------------------
extern "C" void kernel_launch(void* const* d_in, const int* in_sizes, int n_in,
                              void* d_out, int out_size, void* d_ws, size_t ws_size,
                              hipStream_t stream)
{
    const float* x  = (const float*)d_in[0];
    const float* Wq = (const float*)d_in[1];
    const float* Wk = (const float*)d_in[2];
    const float* Wv = (const float*)d_in[3];
    float* out = (float*)d_out;

    char* ws = (char*)d_ws;
    const size_t SZ_X  = (size_t)M_ * D_ * 2;   // 16 MiB
    const size_t SZ_W  = (size_t)D_ * D_ * 2;   //  2 MiB
    const size_t SZ_QK = (size_t)M_ * D_ * 2;   // 16 MiB
    __bf16* Xb  = (__bf16*)(ws);
    __bf16* Wqb = (__bf16*)(ws + SZ_X);
    __bf16* Wkb = (__bf16*)(ws + SZ_X + SZ_W);
    __bf16* Wvb = (__bf16*)(ws + SZ_X + 2 * SZ_W);
    __bf16* Qb  = (__bf16*)(ws + SZ_X + 3 * SZ_W);
    __bf16* Kb  = (__bf16*)(ws + SZ_X + 3 * SZ_W + SZ_QK);
    __bf16* Vt  = (__bf16*)(ws + SZ_X + 3 * SZ_W + 2 * SZ_QK);
    float*  Sc  = (float*) (ws + SZ_X + 3 * SZ_W + 3 * SZ_QK);              // 64 MiB
    __bf16* P   = (__bf16*)(ws + SZ_X + 3 * SZ_W + 3 * SZ_QK
                               + (size_t)M_ * S_ * 4);                      // 32 MiB

    dim3 blk(256);

    // one-shot f32 -> bf16 conversion of x and the three weight matrices
    cvt_bf16_kernel<<<dim3((M_ * D_) / 2048), blk, 0, stream>>>(x,  Xb);
    cvt_bf16_kernel<<<dim3((D_ * D_) / 2048), blk, 0, stream>>>(Wq, Wqb);
    cvt_bf16_kernel<<<dim3((D_ * D_) / 2048), blk, 0, stream>>>(Wk, Wkb);
    cvt_bf16_kernel<<<dim3((D_ * D_) / 2048), blk, 0, stream>>>(Wv, Wvb);

    dim3 gproj(M_ / 128, D_ / 64);                      // 64 x 16
    proj_kernel<<<gproj, blk, 0, stream>>>(Xb, Wqb, Qb, 0);
    proj_kernel<<<gproj, blk, 0, stream>>>(Xb, Wkb, Kb, 0);
    proj_kernel<<<gproj, blk, 0, stream>>>(Xb, Wvb, Vt, 1);   // V stored transposed

    const float scale = 1.0f / ((float)D_ * (float)D_); // reference: / d_out^2
    dim3 gsc(S_ / 128, S_ / 128, B_);                   // 16 x 16 x 4
    scores_kernel<<<gsc, blk, 0, stream>>>(Qb, Kb, Sc, scale);

    softmax_kernel<<<dim3(M_), blk, 0, stream>>>(Sc, P);

    dim3 gctx(S_ / 128, D_ / 64, B_);                   // 16 x 16 x 4
    context_kernel<<<gctx, blk, 0, stream>>>(P, Vt, out);
}
